// NNMambaBlock_86543591014779
// MI455X (gfx1250) — compile-verified, matrix operand order checked by reading, not run on hardware
//
#include <hip/hip_runtime.h>
#include <hip/hip_bf16.h>
#include <math.h>

// ---------------------------------------------------------------------------
// Mamba block for MI455X (gfx1250, wave32, WMMA + TDM).
//   B=4, L=2048, D_MODEL=1024, D_INNER=2048, D_STATE=16, D_CONV=4, DT_RANK=64
// All GEMM operands pre-converted to f16 once; GEMMs run on
// v_wmma_f32_16x16x32_f16 (f32 accumulate) with pure b128 f16 fragment loads.
// Scan stages B/C tiles into LDS via the Tensor Data Mover (6-arg builtin,
// amdgpu-toolchain / clang-23 signature).
// ---------------------------------------------------------------------------

#define B_SZ     4
#define L_LEN    2048
#define D_MODEL  1024
#define D_INNER  2048
#define D_STATE  16
#define D_CONV   4
#define DT_RANK  64
#define E2       (2 * D_INNER)          // 4096
#define XDBL_N   (DT_RANK + 2*D_STATE)  // 96
#define ROWS     (B_SZ * L_LEN)         // 8192

typedef __attribute__((ext_vector_type(16))) _Float16 v16h;
typedef __attribute__((ext_vector_type(8)))  _Float16 v8h;
typedef __attribute__((ext_vector_type(4)))  _Float16 v4h;
typedef __attribute__((ext_vector_type(8)))  float    v8f;
typedef __attribute__((ext_vector_type(4)))  unsigned int u32x4;
typedef __attribute__((ext_vector_type(8)))  int      i32x8;
typedef __attribute__((ext_vector_type(4)))  int      i32x4;

#if defined(__has_builtin)
# if __has_builtin(__builtin_amdgcn_tensor_load_to_lds) && __has_builtin(__builtin_amdgcn_s_wait_tensorcnt)
#  define MAMBA_USE_TDM 1
# endif
#endif

__device__ __forceinline__ float clip01(float v) {
  return __builtin_fminf(__builtin_fmaxf(v, 0.0f), 1.0f);
}

// ---------------------------------------------------------------------------
// f16 fragment loader per ISA 16-bit A/B layout: lane t holds row
// (row0 + t%16); halves 0..7 = K (t<16?0:8)+h, halves 8..15 = 16+(t<16?0:8)+h.
// Two 16-byte b128 loads per fragment, no conversion in the hot loop.
// ---------------------------------------------------------------------------
__device__ __forceinline__ v16h ldfrag(const _Float16* __restrict__ base,
                                       long long ld, int row0, int rmax,
                                       int k, int lane) {
  int r = row0 + (lane & 15);
  r = (r > rmax) ? rmax : r;
  const _Float16* p = base + (long long)r * ld + k + ((lane & 16) ? 8 : 0);
  union { v16h v; v8h h[2]; } u;
  u.h[0] = *(const v8h*)(p);
  u.h[1] = *(const v8h*)(p + 16);
  return u.v;
}

// Epilogues: 0 = none, 1 = clip[0,1], 2 = +bias -> softplus -> clamp[1e-4,20]
template<int EPI>
__device__ __forceinline__ void store_tile(float* __restrict__ C, long long ldc,
                                           const float* __restrict__ bias,
                                           int m0, int n0, int N, v8f c, int lane) {
  int cn = n0 + (lane & 15);
  if (cn >= N) return;
  int rbase = m0 + ((lane >> 4) << 3);
  float bv = (EPI == 2) ? bias[cn] : 0.0f;
#pragma unroll
  for (int v = 0; v < 8; ++v) {
    float val = c[v];
    if (EPI == 1) val = clip01(val);
    if (EPI == 2) {
      val += bv;
      float sp = (val > 20.0f) ? val : log1pf(__expf(val));
      val = __builtin_fminf(__builtin_fmaxf(sp, 1e-4f), 20.0f);
    }
    C[(long long)(rbase + v) * ldc + cn] = val;
  }
}

// ---------------------------------------------------------------------------
// WMMA GEMM: C[M,N] = epi(A[M,K]f16 @ W[N,K]f16^T), f32 out.
// 8 waves/block; wave tile 32x64 (2 A frags x 4 B frags -> 8 v_wmma per
// K-step vs 12 b128 loads). Block tile 128x128. M % 128 == 0; N % 32 == 0.
// ---------------------------------------------------------------------------
template<int EPI>
__global__ __launch_bounds__(256)
void wmma_gemm_f16(const _Float16* __restrict__ A, const _Float16* __restrict__ W,
                   const float* __restrict__ bias, float* __restrict__ C,
                   int M, int N, int K,
                   long long lda, long long ldb, long long ldc) {
  int lane = threadIdx.x & 31;
  int w    = threadIdx.x >> 5;     // 0..7
  int wm   = w & 3;                // 4 waves along M
  int wn   = w >> 2;               // 2 waves along N
  int m0 = blockIdx.y * 128 + wm * 32;
  int n0 = blockIdx.x * 128 + wn * 64;
  if (n0 >= N) return;

  v8f acc[2][4];
#pragma unroll
  for (int i = 0; i < 2; ++i)
#pragma unroll
    for (int j = 0; j < 4; ++j) acc[i][j] = v8f{};

  for (int k = 0; k < K; k += 32) {
    if (k + 32 < K) {  // gfx1250 global_prefetch_b8 of next K-tile
      __builtin_prefetch(A + (long long)(m0 + (lane & 15)) * lda + k + 32, 0, 1);
      __builtin_prefetch(W + (long long)(n0 + (lane & 15)) * ldb + k + 32, 0, 1);
    }
    v16h a0 = ldfrag(A, lda, m0,      M - 1, k, lane);
    v16h a1 = ldfrag(A, lda, m0 + 16, M - 1, k, lane);
    v16h b0 = ldfrag(W, ldb, n0,      N - 1, k, lane);
    v16h b1 = ldfrag(W, ldb, n0 + 16, N - 1, k, lane);
    v16h b2 = ldfrag(W, ldb, n0 + 32, N - 1, k, lane);
    v16h b3 = ldfrag(W, ldb, n0 + 48, N - 1, k, lane);
    acc[0][0] = __builtin_amdgcn_wmma_f32_16x16x32_f16(false, a0, false, b0, (short)0, acc[0][0], false, false);
    acc[0][1] = __builtin_amdgcn_wmma_f32_16x16x32_f16(false, a0, false, b1, (short)0, acc[0][1], false, false);
    acc[0][2] = __builtin_amdgcn_wmma_f32_16x16x32_f16(false, a0, false, b2, (short)0, acc[0][2], false, false);
    acc[0][3] = __builtin_amdgcn_wmma_f32_16x16x32_f16(false, a0, false, b3, (short)0, acc[0][3], false, false);
    acc[1][0] = __builtin_amdgcn_wmma_f32_16x16x32_f16(false, a1, false, b0, (short)0, acc[1][0], false, false);
    acc[1][1] = __builtin_amdgcn_wmma_f32_16x16x32_f16(false, a1, false, b1, (short)0, acc[1][1], false, false);
    acc[1][2] = __builtin_amdgcn_wmma_f32_16x16x32_f16(false, a1, false, b2, (short)0, acc[1][2], false, false);
    acc[1][3] = __builtin_amdgcn_wmma_f32_16x16x32_f16(false, a1, false, b3, (short)0, acc[1][3], false, false);
  }
#pragma unroll
  for (int i = 0; i < 2; ++i)
#pragma unroll
    for (int j = 0; j < 4; ++j)
      store_tile<EPI>(C, ldc, bias, m0 + 16 * i, n0 + 16 * j, N, acc[i][j], lane);
}

// ---------------------------------------------------------------------------
// f32 -> f16 converters (vectorized x4).  n must be a multiple of 4.
// ---------------------------------------------------------------------------
__global__ __launch_bounds__(256)
void cvt_f16_kernel(const float* __restrict__ src, _Float16* __restrict__ dst,
                    long long n4) {
  long long i = (long long)blockIdx.x * 256 + threadIdx.x;
  if (i >= n4) return;
  float4 v = ((const float4*)src)[i];
  v4h h; h[0] = (_Float16)v.x; h[1] = (_Float16)v.y;
  h[2] = (_Float16)v.z; h[3] = (_Float16)v.w;
  ((v4h*)dst)[i] = h;
}

// clip01(x_dbl[:, :64]) -> f16, src ld = 96, dst ld = 64
__global__ __launch_bounds__(256)
void cvt_dtin_kernel(const float* __restrict__ xdbl, _Float16* __restrict__ dst) {
  long long i = (long long)blockIdx.x * 256 + threadIdx.x;   // ROWS*16 quads
  if (i >= (long long)ROWS * 16) return;
  long long row = i >> 4; int c4 = (int)(i & 15);
  float4 v = *(const float4*)(xdbl + row * XDBL_N + c4 * 4);
  v4h h; h[0] = (_Float16)clip01(v.x); h[1] = (_Float16)clip01(v.y);
  h[2] = (_Float16)clip01(v.z); h[3] = (_Float16)clip01(v.w);
  *(v4h*)(dst + row * DT_RANK + c4 * 4) = h;
}

// ---------------------------------------------------------------------------
// Depthwise causal conv (k=4) + bias + silu + clip; writes f32 (scan) + f16
// (GEMM operand).  x lives in xz[..., :D_INNER].
// ---------------------------------------------------------------------------
__global__ __launch_bounds__(256)
void conv_silu_kernel(const float* __restrict__ xz, const float* __restrict__ cw,
                      const float* __restrict__ cb, float* __restrict__ xc,
                      _Float16* __restrict__ xc16) {
  long long i = (long long)blockIdx.x * 256 + threadIdx.x;    // B*L*D_INNER
  if (i >= (long long)ROWS * D_INNER) return;
  int d        = (int)(i & (D_INNER - 1));
  long long bl = i >> 11;
  int l        = (int)(bl & (L_LEN - 1));
  long long b  = bl >> 11;
  float acc = cb[d];
#pragma unroll
  for (int j = 0; j < D_CONV; ++j) {
    int ls = l - (D_CONV - 1) + j;
    if (ls >= 0)
      acc += xz[((b * L_LEN + ls) * (long long)E2) + d] * cw[d * D_CONV + j];
  }
  float s = clip01(acc / (1.0f + __expf(-acc)));   // silu + clip
  xc[i]   = s;
  xc16[i] = (_Float16)s;
}

// ---------------------------------------------------------------------------
// TDM 2D tile load: tile_h rows x tile_w 4-byte elements, row stride
// stride_elems, global -> LDS.  D# per CDNA5 ISA §8.3/8.4; 6-arg builtin
// (clang-23 / therock-10.0): (g0, g1, g2, g3, g_extra, cpol).
// ---------------------------------------------------------------------------
#ifdef MAMBA_USE_TDM
__device__ __forceinline__ void tdm_load_2d(unsigned lds_off, const void* gaddr,
                                            unsigned tile_w, unsigned tile_h,
                                            unsigned stride_elems) {
  unsigned long long ga = (unsigned long long)(size_t)gaddr;
  u32x4 g0;
  g0[0] = 1u;                                         // count=1, user mode
  g0[1] = lds_off;                                    // lds_addr (bytes)
  g0[2] = (unsigned)(ga & 0xFFFFFFFFull);             // global_addr[31:0]
  g0[3] = (unsigned)((ga >> 32) & 0x1FFFFFFull) | (2u << 30); // addr[56:32], type=2
  i32x8 g1;
  g1[0] = (int)(2u << 16);            // data_size=2 (4B), no multicast/pad
  g1[1] = (int)(tile_w << 16);        // tensor_dim0[15:0]  @ bits[63:48]
  g1[2] = (int)(tile_h << 16);        // tensor_dim1[15:0]  @ bits[95:80]
  g1[3] = (int)(tile_w << 16);        // tile_dim0          @ bits[127:112]
  g1[4] = (int)(tile_h & 0xFFFFu);    // tile_dim1          @ bits[143:128]
  g1[5] = (int)stride_elems;          // tensor_dim0_stride[31:0]
  g1[6] = 0;
  g1[7] = 0;
  i32x4 z4 = {};                      // group2/group3: unused (2D tensor)
  i32x8 z8 = {};                      // trailing group (clang-23 form)
  __builtin_amdgcn_tensor_load_to_lds(g0, g1, z4, z4, z8, 0);
}
#endif

// ---------------------------------------------------------------------------
// Selective scan. Block = (b, 256 channels); h[16] per thread in VGPRs.
// B_t/C_t (32 floats/step) staged through LDS in 64-step chunks via TDM.
// Fused: y = clip01(clip01(y + x*D) * silu(z)); writes f16 for the out-GEMM.
// ---------------------------------------------------------------------------
#define CHUNK 64
__global__ __launch_bounds__(256)
void scan_kernel(const float* __restrict__ dtv, const float* __restrict__ xc,
                 const float* __restrict__ xdbl, const float* __restrict__ xz,
                 const float* __restrict__ A_log, const float* __restrict__ Dp,
                 _Float16* __restrict__ y16) {
  __shared__ float sBC[CHUNK][2 * D_STATE];
  int b = blockIdx.y;
  int d = blockIdx.x * 256 + threadIdx.x;

  float Ad[D_STATE];
#pragma unroll
  for (int n = 0; n < D_STATE; ++n) Ad[n] = __expf(A_log[d * D_STATE + n]);
  float Dd = Dp[d];
  float h[D_STATE];
#pragma unroll
  for (int n = 0; n < D_STATE; ++n) h[n] = 0.0f;

  for (int l0 = 0; l0 < L_LEN; l0 += CHUNK) {
    __syncthreads();                       // previous chunk fully consumed
#ifdef MAMBA_USE_TDM
    if (threadIdx.x < 32) {                // wave 0 issues the DMA
      tdm_load_2d((unsigned)(size_t)&sBC[0][0],
                  xdbl + ((long long)(b * L_LEN + l0)) * XDBL_N + DT_RANK,
                  2 * D_STATE, CHUNK, XDBL_N);
    }
    __builtin_amdgcn_s_wait_tensorcnt(0);  // non-issuing waves pass trivially
    __syncthreads();
#else
    for (int i = threadIdx.x; i < CHUNK * 2 * D_STATE; i += 256) {
      int step = i >> 5, j = i & 31;
      sBC[step][j] =
          xdbl[((long long)(b * L_LEN + l0 + step)) * XDBL_N + DT_RANK + j];
    }
    __syncthreads();
#endif
    for (int s = 0; s < CHUNK; ++s) {
      int l = l0 + s;
      long long idx = ((long long)b * L_LEN + l) * D_INNER + d;
      float dt_t = dtv[idx];
      float x_t  = xc[idx];
      float du   = dt_t * x_t;
      float yv   = 0.0f;
#pragma unroll
      for (int n = 0; n < D_STATE; ++n) {
        float dA = __expf(-dt_t * Ad[n]);
        h[n] = fmaf(dA, h[n], du * sBC[s][n]);
        yv   = fmaf(h[n], sBC[s][D_STATE + n], yv);
      }
      float z = xz[((long long)(b * L_LEN + l)) * E2 + D_INNER + d];
      yv = clip01(yv + x_t * Dd);
      float gate = z / (1.0f + __expf(-z));
      y16[idx] = (_Float16)clip01(yv * gate);
    }
  }
}

// ---------------------------------------------------------------------------
extern "C" void kernel_launch(void* const* d_in, const int* in_sizes, int n_in,
                              void* d_out, int out_size, void* d_ws, size_t ws_size,
                              hipStream_t stream) {
  const float* hs    = (const float*)d_in[0];
  const float* W_in  = (const float*)d_in[1];
  const float* convw = (const float*)d_in[2];
  const float* convb = (const float*)d_in[3];
  const float* W_x   = (const float*)d_in[4];
  const float* W_dt  = (const float*)d_in[5];
  const float* b_dt  = (const float*)d_in[6];
  const float* W_out = (const float*)d_in[7];
  const float* A_log = (const float*)d_in[8];
  const float* Dp    = (const float*)d_in[9];
  float* out = (float*)d_out;

  // Workspace layout: f32 intermediates, then f16 operands (~370 MB).
  char* p = (char*)d_ws;
  float* xz   = (float*)p;  p += (long long)ROWS * E2      * 4;
  float* xc   = (float*)p;  p += (long long)ROWS * D_INNER * 4;
  float* xdbl = (float*)p;  p += (long long)ROWS * XDBL_N  * 4;
  float* dt   = (float*)p;  p += (long long)ROWS * D_INNER * 4;
  _Float16* hs16   = (_Float16*)p;  p += (long long)ROWS * D_MODEL     * 2;
  _Float16* Win16  = (_Float16*)p;  p += (long long)E2 * D_MODEL       * 2;
  _Float16* Wx16   = (_Float16*)p;  p += (long long)XDBL_N * D_INNER   * 2;
  _Float16* Wdt16  = (_Float16*)p;  p += (long long)D_INNER * DT_RANK  * 2;
  _Float16* Wout16 = (_Float16*)p;  p += (long long)D_MODEL * D_INNER  * 2;
  _Float16* xc16   = (_Float16*)p;  p += (long long)ROWS * D_INNER     * 2;
  _Float16* dtin16 = (_Float16*)p;  p += (long long)ROWS * DT_RANK     * 2;
  _Float16* y16    = (_Float16*)p;  p += (long long)ROWS * D_INNER     * 2;

  dim3 blk(256);
  auto cvt = [&](const float* s, _Float16* d, long long n) {
    long long n4 = n / 4;
    cvt_f16_kernel<<<(unsigned)((n4 + 255) / 256), blk, 0, stream>>>(s, d, n4);
  };

  // 0) one-time f32 -> f16 operand conversions
  cvt(hs,    hs16,   (long long)ROWS * D_MODEL);
  cvt(W_in,  Win16,  (long long)E2 * D_MODEL);
  cvt(W_x,   Wx16,   (long long)XDBL_N * D_INNER);
  cvt(W_dt,  Wdt16,  (long long)D_INNER * DT_RANK);
  cvt(W_out, Wout16, (long long)D_MODEL * D_INNER);

  // 1) xz = clip01(hs @ W_in^T)                       M=8192 N=4096 K=1024
  wmma_gemm_f16<1><<<dim3(E2 / 128, ROWS / 128), blk, 0, stream>>>(
      hs16, Win16, nullptr, xz, ROWS, E2, D_MODEL, D_MODEL, D_MODEL, E2);

  // 2) depthwise conv + silu + clip -> xc (f32) + xc16 (f16)
  long long nconv = (long long)ROWS * D_INNER;
  conv_silu_kernel<<<(unsigned)((nconv + 255) / 256), blk, 0, stream>>>(
      xz, convw, convb, xc, xc16);

  // 3) x_dbl = xc @ W_x^T                             M=8192 N=96 K=2048
  wmma_gemm_f16<0><<<dim3((XDBL_N + 127) / 128, ROWS / 128), blk, 0, stream>>>(
      xc16, Wx16, nullptr, xdbl, ROWS, XDBL_N, D_INNER, D_INNER, D_INNER, XDBL_N);

  // 3b) dtin16 = f16(clip01(x_dbl[:, :64]))
  cvt_dtin_kernel<<<(unsigned)((ROWS * 16 + 255) / 256), blk, 0, stream>>>(
      xdbl, dtin16);

  // 4) dt = clamp(softplus(dtin @ W_dt^T + b_dt))     M=8192 N=2048 K=64
  wmma_gemm_f16<2><<<dim3(D_INNER / 128, ROWS / 128), blk, 0, stream>>>(
      dtin16, Wdt16, b_dt, dt, ROWS, D_INNER, DT_RANK, DT_RANK, DT_RANK, D_INNER);

  // 5) selective scan (TDM-staged B/C) -> y16
  scan_kernel<<<dim3(D_INNER / 256, B_SZ), blk, 0, stream>>>(
      dt, xc, xdbl, xz, A_log, Dp, y16);

  // 6) out = y @ W_out^T                              M=8192 N=1024 K=2048
  wmma_gemm_f16<0><<<dim3(D_MODEL / 128, ROWS / 128), blk, 0, stream>>>(
      y16, Wout16, nullptr, out, ROWS, D_MODEL, D_INNER, D_INNER, D_INNER, D_MODEL);
}
